// GroupedMultiQueryAttention_1614907704000
// MI455X (gfx1250) — matrix operand order repaired and checked by reading, not compiled, not run on hardware
//
#include <hip/hip_runtime.h>

// GroupedMultiQueryAttention — MI455X (gfx1250, wave32, WMMA)
//
// The reference's einsums contract attention's q/k axes away entirely:
//   attention[g,b,s,h,d] = (sum_{q,k} softmax(energy)[q,k]) * v[g,b,s,h,d]
//                        = HEAD_DIM * v            (softmax rows each sum to 1)
// so   out = tensor @ (Wv @ WoSum),  WoSum[kv*64+d,o] = 64 * sum_r Wo[(4kv+r)*64+d, o].
// Wq/Wk/RoPE/softmax cancel mathematically (delta ~1e-7 relative, below fp32 GEMM noise).
//
// Roofline: 68.7 GFLOP (f16 WMMA, f32 accum) vs ~310 MB HBM traffic (13.3 us at
// 23.3 TB/s) -> balanced; per-genome x (16.8 MB) and Weff (2 MB f16) are L2-resident.
//
// Stage 1: Weff[g] (1024x1024, f16) = Wv[g] (1024x256) @ WoSum[g]      [64x64 tile]
// Stage 2: out[g]  (4096x1024, f32) = tensor[g] (->f16) @ Weff[g]      [128x128 tile,
//          256 thr / 8 waves, register-pipelined global->LDS staging]

#define GENOMES 8
#define BATCH 2
#define SEQ 2048
#define EMBED 1024
#define HEADS 16
#define KV_HEADS 4
#define HEAD_DIM 64

typedef __attribute__((ext_vector_type(16))) _Float16 v16h;
typedef __attribute__((ext_vector_type(8)))  float    v8f;

union H16 { v16h v; _Float16 h[16]; };

// A-matrix (16x32 f16) per-lane K index for VGPR-slot v (ISA 7.12.2):
// v in 0..3 -> K = 8*half + 2v ; v in 4..7 -> K = 16 + 8*half + 2(v-4)
__device__ __forceinline__ int a_kmap(int v, int half) {
  return (v < 4) ? (8 * half + 2 * v) : (16 + 8 * half + 2 * (v - 4));
}

__device__ __forceinline__ unsigned int pkh(float a, float b) {
  _Float16 ha = (_Float16)a, hb = (_Float16)b;
  return (unsigned int)__builtin_bit_cast(unsigned short, ha) |
         ((unsigned int)__builtin_bit_cast(unsigned short, hb) << 16);
}

// ---------------------------------------------------------------------------
// Stage 1: Weff[g] = Wv[g] @ (64 * GQA-summed Wo[g]).  M=1024, N=1024, K=256.
// B operand is formed on the fly while staging: 4 Wo rows summed in f32.
// ---------------------------------------------------------------------------
__global__ void __launch_bounds__(128)
build_weff_kernel(const float* __restrict__ Wv, const float* __restrict__ Wo,
                  unsigned short* __restrict__ Weff) {
  __shared__ _Float16 As[64][34];
  __shared__ _Float16 Bs[32][66];

  const int g = blockIdx.z;
  const float* Ag  = Wv + (size_t)g * EMBED * (KV_HEADS * HEAD_DIM);
  const float* WoG = Wo + (size_t)g * EMBED * EMBED;
  unsigned short* Cg = Weff + (size_t)g * EMBED * EMBED;

  const int m0 = blockIdx.x * 64;
  const int n0 = blockIdx.y * 64;
  const int tid  = threadIdx.x;
  const int lane = tid & 31, wid = tid >> 5;
  const int half = lane >> 4, l15 = lane & 15;
  const int wm = (wid >> 1) * 32, wn = (wid & 1) * 32;

  const v8f vzero = {0.f, 0.f, 0.f, 0.f, 0.f, 0.f, 0.f, 0.f};
  v8f acc[2][2];
  acc[0][0] = vzero; acc[0][1] = vzero; acc[1][0] = vzero; acc[1][1] = vzero;

  const int Kdim = KV_HEADS * HEAD_DIM;  // 256
  for (int k0 = 0; k0 < Kdim; k0 += 32) {
#pragma unroll
    for (int i = 0; i < 16; ++i) {
      int e = tid + 128 * i;
      int r = e >> 5, c = e & 31;
      As[r][c] = (_Float16)Ag[(size_t)(m0 + r) * Kdim + (k0 + c)];
    }
#pragma unroll
    for (int i = 0; i < 16; ++i) {
      int e = tid + 128 * i;
      int r = e >> 6, c = e & 63;
      int j  = k0 + r;                // reduced index 0..255
      int kv = j >> 6, d = j & 63;
      const float* wrp = WoG + (size_t)(kv * 4 * HEAD_DIM + d) * EMBED + (n0 + c);
      float s = wrp[0]
              + wrp[(size_t)(1 * HEAD_DIM) * EMBED]
              + wrp[(size_t)(2 * HEAD_DIM) * EMBED]
              + wrp[(size_t)(3 * HEAD_DIM) * EMBED];
      Bs[r][c] = (_Float16)(64.0f * s);
    }
    __syncthreads();

    H16 af[2], bfr[2];
#pragma unroll
    for (int t = 0; t < 2; ++t) {
      int row = wm + 16 * t + l15;
#pragma unroll
      for (int v = 0; v < 8; ++v) {
        int kk = a_kmap(v, half);
        af[t].h[2 * v]     = As[row][kk];
        af[t].h[2 * v + 1] = As[row][kk + 1];
      }
    }
#pragma unroll
    for (int t = 0; t < 2; ++t) {
      int col = wn + 16 * t + l15;
#pragma unroll
      for (int j = 0; j < 8; ++j) {
        bfr[t].h[2 * j]     = Bs[a_kmap(j, half)][col];
        bfr[t].h[2 * j + 1] = Bs[a_kmap(j, half) + 1][col];
      }
    }
#pragma unroll
    for (int i = 0; i < 2; ++i)
#pragma unroll
      for (int j = 0; j < 2; ++j)
        acc[i][j] = __builtin_amdgcn_wmma_f32_16x16x32_f16(
            false, af[i].v, false, bfr[j].v, (short)0, acc[i][j], false, false);
    __syncthreads();
  }

#pragma unroll
  for (int i = 0; i < 2; ++i)
#pragma unroll
    for (int j = 0; j < 2; ++j)
#pragma unroll
      for (int r = 0; r < 8; ++r) {
        int m = m0 + wm + 16 * i + r + 8 * half;
        int n = n0 + wn + 16 * j + l15;
        _Float16 hv = (_Float16)acc[i][j][r];
        Cg[(size_t)m * EMBED + n] = __builtin_bit_cast(unsigned short, hv);
      }
}

// ---------------------------------------------------------------------------
// Stage 2: out[g] = tensor[g] (4096x1024 f32 -> f16) @ Weff[g] (1024x1024 f16)
// 128x128 tile, 256 threads (8 waves, 4x2), register-pipelined staging.
// ---------------------------------------------------------------------------
__global__ void __launch_bounds__(256)
xweff_gemm_kernel(const float* __restrict__ X, const unsigned short* __restrict__ Weff,
                  float* __restrict__ Out) {
  __shared__ _Float16 As[128][34];    // 128 x 32 (+pad), f16
  __shared__ _Float16 Bs[32][130];    // 32 x 128 (+pad), f16

  const int g = blockIdx.z;
  const float* Ag = X + (size_t)g * (BATCH * SEQ) * EMBED;
  const unsigned short* Bg = Weff + (size_t)g * EMBED * EMBED;
  float* Cg = Out + (size_t)g * (BATCH * SEQ) * EMBED;

  const int m0 = blockIdx.x * 128;
  const int n0 = blockIdx.y * 128;
  const int tid  = threadIdx.x;
  const int lane = tid & 31, wid = tid >> 5;
  const int half = lane >> 4, l15 = lane & 15;
  const int wr = (wid >> 1) * 32;     // wave row offset (4 row-groups)
  const int wc = (wid & 1) * 64;      // wave col offset (2 col-groups)

  // staging coordinates: A thread -> (row, 16 contiguous f32), B thread -> (row, 16 halves)
  const int arow = tid >> 1, acol = (tid & 1) * 16;
  const int brow = tid >> 3, bcol = (tid & 7) * 16;

  const v8f vzero = {0.f, 0.f, 0.f, 0.f, 0.f, 0.f, 0.f, 0.f};
  v8f acc[2][4];
#pragma unroll
  for (int i = 0; i < 2; ++i)
#pragma unroll
    for (int j = 0; j < 4; ++j) acc[i][j] = vzero;

  float4 ra0, ra1, ra2, ra3;
  uint4  rb0, rb1;

  // ---- prologue: load K-tile 0 into registers, commit to LDS
  {
    const float4* ap = reinterpret_cast<const float4*>(Ag + (size_t)(m0 + arow) * EMBED + acol);
    ra0 = ap[0]; ra1 = ap[1]; ra2 = ap[2]; ra3 = ap[3];
    const uint4* bp = reinterpret_cast<const uint4*>(Bg + (size_t)brow * EMBED + n0 + bcol);
    rb0 = bp[0]; rb1 = bp[1];
    unsigned int* ad = reinterpret_cast<unsigned int*>(&As[arow][acol]);
    ad[0] = pkh(ra0.x, ra0.y); ad[1] = pkh(ra0.z, ra0.w);
    ad[2] = pkh(ra1.x, ra1.y); ad[3] = pkh(ra1.z, ra1.w);
    ad[4] = pkh(ra2.x, ra2.y); ad[5] = pkh(ra2.z, ra2.w);
    ad[6] = pkh(ra3.x, ra3.y); ad[7] = pkh(ra3.z, ra3.w);
    unsigned int* bd = reinterpret_cast<unsigned int*>(&Bs[brow][bcol]);
    bd[0] = rb0.x; bd[1] = rb0.y; bd[2] = rb0.z; bd[3] = rb0.w;
    bd[4] = rb1.x; bd[5] = rb1.y; bd[6] = rb1.z; bd[7] = rb1.w;
  }
  __syncthreads();

  for (int k0 = 0; k0 < EMBED; k0 += 32) {
    const bool more = (k0 + 32) < EMBED;
    // ---- issue next tile's global loads (hidden behind WMMA work)
    if (more) {
      const float4* ap = reinterpret_cast<const float4*>(
          Ag + (size_t)(m0 + arow) * EMBED + (k0 + 32) + acol);
      ra0 = ap[0]; ra1 = ap[1]; ra2 = ap[2]; ra3 = ap[3];
      const uint4* bp = reinterpret_cast<const uint4*>(
          Bg + (size_t)(k0 + 32 + brow) * EMBED + n0 + bcol);
      rb0 = bp[0]; rb1 = bp[1];
    }
    if (k0 + 64 < EMBED) {  // keep a prefetch 2 tiles ahead (global_prefetch_b8)
      __builtin_prefetch(Ag + (size_t)(m0 + arow) * EMBED + (k0 + 64) + acol, 0, 0);
    }

    // ---- compute on current LDS tile
    H16 af[2], bfr[4];
#pragma unroll
    for (int t = 0; t < 2; ++t) {
      int row = wr + 16 * t + l15;
#pragma unroll
      for (int v = 0; v < 8; ++v) {
        int kk = a_kmap(v, half);
        af[t].h[2 * v]     = As[row][kk];
        af[t].h[2 * v + 1] = As[row][kk + 1];
      }
    }
#pragma unroll
    for (int t = 0; t < 4; ++t) {
      int col = wc + 16 * t + l15;
#pragma unroll
      for (int j = 0; j < 8; ++j) {
        bfr[t].h[2 * j]     = Bs[a_kmap(j, half)][col];
        bfr[t].h[2 * j + 1] = Bs[a_kmap(j, half) + 1][col];
      }
    }
#pragma unroll
    for (int i = 0; i < 2; ++i)
#pragma unroll
      for (int j = 0; j < 4; ++j)
        acc[i][j] = __builtin_amdgcn_wmma_f32_16x16x32_f16(
            false, af[i].v, false, bfr[j].v, (short)0, acc[i][j], false, false);

    __syncthreads();              // everyone done reading current tile
    if (more) {                   // commit next tile regs -> LDS
      unsigned int* ad = reinterpret_cast<unsigned int*>(&As[arow][acol]);
      ad[0] = pkh(ra0.x, ra0.y); ad[1] = pkh(ra0.z, ra0.w);
      ad[2] = pkh(ra1.x, ra1.y); ad[3] = pkh(ra1.z, ra1.w);
      ad[4] = pkh(ra2.x, ra2.y); ad[5] = pkh(ra2.z, ra2.w);
      ad[6] = pkh(ra3.x, ra3.y); ad[7] = pkh(ra3.z, ra3.w);
      unsigned int* bd = reinterpret_cast<unsigned int*>(&Bs[brow][bcol]);
      bd[0] = rb0.x; bd[1] = rb0.y; bd[2] = rb0.z; bd[3] = rb0.w;
      bd[4] = rb1.x; bd[5] = rb1.y; bd[6] = rb1.z; bd[7] = rb1.w;
    }
    __syncthreads();              // next tile visible
  }

#pragma unroll
  for (int i = 0; i < 2; ++i)
#pragma unroll
    for (int j = 0; j < 4; ++j)
#pragma unroll
      for (int r = 0; r < 8; ++r) {
        int m = m0 + wr + 16 * i + r + 8 * half;
        int n = n0 + wc + 16 * j + l15;
        Cg[(size_t)m * EMBED + n] = acc[i][j][r];
      }
}

// ---------------------------------------------------------------------------

extern "C" void kernel_launch(void* const* d_in, const int* in_sizes, int n_in,
                              void* d_out, int out_size, void* d_ws, size_t ws_size,
                              hipStream_t stream) {
  (void)in_sizes; (void)n_in; (void)out_size; (void)ws_size;
  const float* tensor = (const float*)d_in[0];
  // d_in[1] = Wq, d_in[2] = Wk: they cancel exactly (softmax rows sum to 1,
  // and the reference einsum sums the whole score matrix -> constant HEAD_DIM).
  const float* Wv = (const float*)d_in[3];
  const float* Wo = (const float*)d_in[4];
  float* out = (float*)d_out;

  // Workspace: Weff f16, GENOMES*1024*1024 halves = 16 MiB
  unsigned short* Weff = (unsigned short*)d_ws;

  dim3 gA(EMBED / 64, EMBED / 64, GENOMES);             // 16 x 16 x 8, 128 thr
  build_weff_kernel<<<gA, dim3(128), 0, stream>>>(Wv, Wo, Weff);

  dim3 gB((BATCH * SEQ) / 128, EMBED / 128, GENOMES);   // 32 x 8 x 8, 256 thr
  xweff_gemm_kernel<<<gB, dim3(256), 0, stream>>>(tensor, Weff, out);
}
